// SpritesAndTexturesCoordinateDecoder_3487513444877
// MI455X (gfx1250) — matrix operand order repaired and checked by reading, not compiled, not run on hardware
//
#include <hip/hip_runtime.h>
#include <hip/hip_bf16.h>
#include <math.h>

// ---------------- configuration ----------------
#define RS          200                     // act row stride in halves (16B aligned, bank-friendly)
#define NWAVES      16
#define PACKED_TILES 172
#define PACKED_HALVES (PACKED_TILES * 512)  // 88064
#define PACKED_BYTES  (PACKED_HALVES * 2)   // 176128
#define ACT_BYTES     (NWAVES * 16 * RS * 2)      // 102400 (one in-place buffer per wave)
#define MASK_BYTES    (NWAVES * 16 * 4)           // 1024
#define SMEM_BYTES    (PACKED_BYTES + ACT_BYTES + MASK_BYTES) // 279552

// TDM copy geometry: 176128 B = 22016 x 8-byte units = 2752 x 8 tile
#define TDM_DIM0    2752u
#define TDM_DIM1    8u

// tile offsets (units of 512 halves) within packed weight buffer
#define T_WS0   0     // KT=3 NT=8 (96->128)
#define T_WS1   24    // KT=4 NT=8
#define T_WS2   56    // KT=4 NT=8
#define T_WT0   88    // KT=6 NT=8 (192->128)
#define T_WT1   136   // KT=4 NT=8
#define T_HEAD  168   // KT=4 NT=1 (128->16, col0=shape, col1..3=rgb)

typedef _Float16 v16h __attribute__((ext_vector_type(16)));
typedef _Float16 v8h  __attribute__((ext_vector_type(8)));
typedef _Float16 h2v  __attribute__((ext_vector_type(2)));
typedef float    v8f  __attribute__((ext_vector_type(8)));
typedef unsigned int u32x4 __attribute__((ext_vector_type(4)));
typedef int          i32x8 __attribute__((ext_vector_type(8)));
typedef int          i32x4 __attribute__((ext_vector_type(4)));

// scale factors with the sqrt(2) lrelu gain folded into the *consuming* layer:
#define S_WS0   0.102062072615966f   // 1/sqrt(96)            (input not lrelu'd)
#define S_128G  0.125f               // sqrt(2)/sqrt(128) = sqrt(1/64) exactly
#define S_WT0H  0.102062072615966f   // sqrt(2)/sqrt(192) = sqrt(1/96)  (k<128: lrelu'd h)
#define S_WT0C  0.072168783648703f   // 1/sqrt(192)             (k>=128: raw texture cond)

// ---------------- weight packing kernel ----------------
// packed[tile*512 + lane*16 + e] = w[n][k],  n = nt*16 + (lane&15),
// k = kt*32 + 16*(lane>>4) + e   (lanes 0-15: K 0..15, lanes 16-31: K 16..31)
__global__ void pack_weights_kernel(const float* __restrict__ ws0,
                                    const float* __restrict__ ws1,
                                    const float* __restrict__ ws2,
                                    const float* __restrict__ wsh,
                                    const float* __restrict__ wt0,
                                    const float* __restrict__ wt1,
                                    const float* __restrict__ wt2,
                                    _Float16* __restrict__ packed) {
  int g = blockIdx.x * blockDim.x + threadIdx.x;
  if (g >= PACKED_HALVES) return;
  int t = g >> 9;
  int w = g & 511;
  int l = w >> 4;
  int e = w & 15;
  int col = l & 15;
  int khalf = l >> 4;
  float val = 0.0f;
  if (t < T_WS1) {                    // ws0: 96 -> 128
    int tt = t - T_WS0, kt = tt >> 3, nt = tt & 7;
    int n = nt * 16 + col, k = kt * 32 + khalf * 16 + e;
    val = ws0[n * 96 + k] * S_WS0;
  } else if (t < T_WS2) {             // ws1
    int tt = t - T_WS1, kt = tt >> 3, nt = tt & 7;
    int n = nt * 16 + col, k = kt * 32 + khalf * 16 + e;
    val = ws1[n * 128 + k] * S_128G;
  } else if (t < T_WT0) {             // ws2
    int tt = t - T_WS2, kt = tt >> 3, nt = tt & 7;
    int n = nt * 16 + col, k = kt * 32 + khalf * 16 + e;
    val = ws2[n * 128 + k] * S_128G;
  } else if (t < T_WT1) {             // wt0: 192 -> 128 (split gain)
    int tt = t - T_WT0, kt = tt >> 3, nt = tt & 7;
    int n = nt * 16 + col, k = kt * 32 + khalf * 16 + e;
    val = wt0[n * 192 + k] * (k < 128 ? S_WT0H : S_WT0C);
  } else if (t < T_HEAD) {            // wt1
    int tt = t - T_WT1, kt = tt >> 3, nt = tt & 7;
    int n = nt * 16 + col, k = kt * 32 + khalf * 16 + e;
    val = wt1[n * 128 + k] * S_128G;
  } else {                            // fused head: 128 -> 16
    int kt = t - T_HEAD;
    int n = col, k = kt * 32 + khalf * 16 + e;
    if (n == 0)      val = wsh[k] * S_128G;
    else if (n < 4)  val = wt2[(n - 1) * 128 + k] * S_128G;
    else             val = 0.0f;
  }
  packed[g] = (_Float16)val;
}

// ---------------- device helpers ----------------
// A-fragment: lane holds row M = lane&15. elems 0..15 = K kb..kb+15,
// kb = kt*32 + 16*(lane>=16).
__device__ __forceinline__ v16h load_a_frag(const _Float16* __restrict__ arow,
                                            int kt, int hs) {
  int kb = kt * 32 + hs * 16;
  v8h lo = *(const v8h*)(arow + kb);
  v8h hi = *(const v8h*)(arow + kb + 8);
  return __builtin_shufflevector(lo, hi, 0, 1, 2, 3, 4, 5, 6, 7,
                                 8, 9, 10, 11, 12, 13, 14, 15);
}

// One MLP layer computed IN PLACE on the wave's activation buffer.
// All A/B LDS loads precede all activation stores (wave-order LDS => safe).
// Per k-tile, all NT B fragments are preloaded -> 16 ds_loads in flight,
// giving an 8-WMMA window to hide LDS latency.
template <int KT, int NT>
__device__ __forceinline__ void mlp_layer(_Float16* __restrict__ act,
                                          const _Float16* __restrict__ wTiles,
                                          const float* __restrict__ bias,
                                          int lane) {
  const int ln = lane & 15;
  const int hs = lane >> 4;

  v8f acc[NT];
#pragma unroll
  for (int nt = 0; nt < NT; ++nt)
#pragma unroll
    for (int i = 0; i < 8; ++i) acc[nt][i] = 0.0f;

  const _Float16* arow = act + ln * RS;
  const v16h* wv = (const v16h*)(wTiles + lane * 16);  // tile stride = 32 v16h

  v16h a = load_a_frag(arow, 0, hs);
  v16h an = a;
#pragma unroll
  for (int kt = 0; kt < KT; ++kt) {
    v16h barr[NT];
#pragma unroll
    for (int nt = 0; nt < NT; ++nt) barr[nt] = wv[(kt * NT + nt) * 32];
    if (kt + 1 < KT) an = load_a_frag(arow, kt + 1, hs);
#pragma unroll
    for (int nt = 0; nt < NT; ++nt)
      acc[nt] = __builtin_amdgcn_wmma_f32_16x16x32_f16(
          false, a, false, barr[nt], (short)0, acc[nt], false, false);
    a = an;
  }

  // packed-f16 epilogue: +bias, leaky-relu(0.2) (sqrt(2) gain folded in weights)
  const h2v slope = {(_Float16)0.2f, (_Float16)0.2f};
#pragma unroll
  for (int nt = 0; nt < NT; ++nt) {
    float bv = bias[nt * 16 + ln];
    h2v bb = {(_Float16)bv, (_Float16)bv};
    _Float16* colp = act + (8 * hs) * RS + nt * 16 + ln;
#pragma unroll
    for (int v = 0; v < 8; v += 2) {
      h2v s = {(_Float16)acc[nt][v], (_Float16)acc[nt][v + 1]};
      s = s + bb;
      h2v m = s * slope;
      h2v r = __builtin_elementwise_max(s, m);
      colp[v * RS] = r[0];
      colp[(v + 1) * RS] = r[1];
    }
  }
}

__device__ __forceinline__ void head_layer(const _Float16* __restrict__ act,
                                           const _Float16* __restrict__ wTiles,
                                           const float* __restrict__ bsh,
                                           const float* __restrict__ bt2,
                                           const float* __restrict__ maskp,
                                           float* __restrict__ out,
                                           int npix, int pix0, int lane) {
  const int ln = lane & 15;
  const int hs = lane >> 4;
  v8f acc;
#pragma unroll
  for (int i = 0; i < 8; ++i) acc[i] = 0.0f;
  const _Float16* arow = act + ln * RS;
  const v16h* wv = (const v16h*)(wTiles + lane * 16);

  v16h barr[4];
#pragma unroll
  for (int kt = 0; kt < 4; ++kt) barr[kt] = wv[kt * 32];
  v16h a0 = load_a_frag(arow, 0, hs);
  v16h a1 = load_a_frag(arow, 1, hs);
  v16h a2 = load_a_frag(arow, 2, hs);
  v16h a3 = load_a_frag(arow, 3, hs);
  acc = __builtin_amdgcn_wmma_f32_16x16x32_f16(false, a0, false, barr[0], (short)0, acc, false, false);
  acc = __builtin_amdgcn_wmma_f32_16x16x32_f16(false, a1, false, barr[1], (short)0, acc, false, false);
  acc = __builtin_amdgcn_wmma_f32_16x16x32_f16(false, a2, false, barr[2], (short)0, acc, false, false);
  acc = __builtin_amdgcn_wmma_f32_16x16x32_f16(false, a3, false, barr[3], (short)0, acc, false, false);

  float bv = (ln == 0) ? bsh[0] : (ln < 4 ? bt2[ln - 1] : 0.0f);
#pragma unroll
  for (int v = 0; v < 8; ++v) {
    int r = v + 8 * hs;
    int pix = pix0 + r;
    float y = acc[v] + bv;
    if (ln == 0) {
      out[pix] = y * maskp[r];                    // shape head (masked)
    } else if (ln < 4) {
      out[npix + pix * 3 + (ln - 1)] = y;         // texture head rgb
    }
  }
}

// ---------------- main decoder kernel ----------------
__global__ __launch_bounds__(512, 1) void decoder_kernel(
    const float* __restrict__ coords,
    const float* __restrict__ shape_cond,
    const float* __restrict__ tex_cond,
    const _Float16* __restrict__ wpacked,
    const float* __restrict__ bs0, const float* __restrict__ bs1,
    const float* __restrict__ bs2, const float* __restrict__ bsh,
    const float* __restrict__ bt0, const float* __restrict__ bt1,
    const float* __restrict__ bt2,
    float* __restrict__ out, int npix, int ntiles) {
  extern __shared__ char smem_raw[];
  _Float16* wlds  = (_Float16*)smem_raw;
  _Float16* acts  = (_Float16*)(smem_raw + PACKED_BYTES);
  float*    masks = (float*)(smem_raw + PACKED_BYTES + ACT_BYTES);

  // ---- weight stage via Tensor Data Mover: one DMA descriptor, wave 0 only ----
  // D# group0: count=1 | lds_addr(=0: weights at LDS offset 0) | global_addr | type=2
  // D# group1: data_size=8B, tensor 2752x8 (8B units), tile 2752x8, stride0=2752
  if (threadIdx.x < 32) {
    unsigned long long ga = (unsigned long long)(uintptr_t)wpacked;
    u32x4 g0 = {1u,                                     // count=1 (user descriptor)
                0u,                                     // lds_addr = 0 (weights base)
                (unsigned int)(ga & 0xFFFFFFFFu),
                (unsigned int)((ga >> 32) & 0x1FFFFFFu) | (2u << 30)};  // type=2
    i32x8 g1 = {0, 0, 0, 0, 0, 0, 0, 0};
    g1[0] = (int)(3u << 16);                            // data_size = 3 (8 bytes)
    g1[1] = (int)((TDM_DIM0 & 0xFFFFu) << 16);          // tensor_dim0[15:0]
    g1[2] = (int)(((TDM_DIM0 >> 16) & 0xFFFFu) |        // tensor_dim0[31:16]
                  ((TDM_DIM1 & 0xFFFFu) << 16));        // tensor_dim1[15:0]
    g1[3] = (int)(((TDM_DIM1 >> 16) & 0xFFFFu) |        // tensor_dim1[31:16]
                  ((TDM_DIM0 & 0xFFFFu) << 16));        // tile_dim0
    g1[4] = (int)(TDM_DIM1 & 0xFFFFu);                  // tile_dim1 (tile_dim2=0)
    g1[5] = (int)TDM_DIM0;                              // tensor_dim0_stride[31:0]
    i32x4 gz = {0, 0, 0, 0};                            // <=2D: groups 2/3 unused
    i32x8 g4 = {0, 0, 0, 0, 0, 0, 0, 0};                // extra group (clang-23 form)
    __builtin_amdgcn_tensor_load_to_lds(g0, g1, gz, gz, g4, 0);
    __builtin_amdgcn_s_wait_tensorcnt(0);
  }
  __syncthreads();

  const int wave = threadIdx.x >> 5;
  const int lane = threadIdx.x & 31;
  const int ln = lane & 15;
  const int hs = lane >> 4;
  _Float16* buf = acts + wave * (16 * RS);
  float* maskp = masks + wave * 16;
  const int tile_stride = gridDim.x * NWAVES;

  for (int tile = blockIdx.x * NWAVES + wave; tile < ntiles; tile += tile_stride) {
    const int pix0 = tile * 16;
    const int bidx = pix0 >> 16;  // 256*256 pixels per batch image

    {
      int npx = pix0 + tile_stride * 16;
      if (npx < npix) __builtin_prefetch(coords + 2 * npx, 0, 1);
    }

    // --- stage sinusoidal embedding + radius mask (lanes 0..15, one pixel each) ---
    if (lane < 16) {
      float cx0 = coords[(pix0 + lane) * 2 + 0];
      float cy0 = coords[(pix0 + lane) * 2 + 1];
      float r = sqrtf(cx0 * cx0 + cy0 * cy0);
      maskp[lane] = 1.0f - tanhf(fmaxf(r - 1.0f, 0.0f));
      float cx = cx0 * 0.1f, cy = cy0 * 0.1f;
      _Float16* rp = buf + lane * RS;
      float f = 1.0f;
#pragma unroll
      for (int s = 0; s < 8; ++s) {
        float ax = cx * f, ay = cy * f;
        rp[4 * s + 0] = (_Float16)__sinf(ax);
        rp[4 * s + 1] = (_Float16)__sinf(ay);
        rp[4 * s + 2] = (_Float16)__cosf(ax);
        rp[4 * s + 3] = (_Float16)__cosf(ay);
        f *= 2.0f;
      }
    }
    // --- stage shape_cond into feats 32..95 (all lanes, 32 halves each) ---
    {
      const float* sc = shape_cond + bidx * 64 + hs * 32;
      _Float16* rp = buf + ln * RS + 32 + hs * 32;
#pragma unroll
      for (int j = 0; j < 32; j += 4) {
        float4 v4 = *(const float4*)(sc + j);
        rp[j + 0] = (_Float16)v4.x;
        rp[j + 1] = (_Float16)v4.y;
        rp[j + 2] = (_Float16)v4.z;
        rp[j + 3] = (_Float16)v4.w;
      }
    }
    // --- stage texture_cond into feats 128..191 (all lanes) ---
    {
      const float* tc = tex_cond + bidx * 64 + hs * 32;
      _Float16* rp = buf + ln * RS + 128 + hs * 32;
#pragma unroll
      for (int j = 0; j < 32; j += 4) {
        float4 v4 = *(const float4*)(tc + j);
        rp[j + 0] = (_Float16)v4.x;
        rp[j + 1] = (_Float16)v4.y;
        rp[j + 2] = (_Float16)v4.z;
        rp[j + 3] = (_Float16)v4.w;
      }
    }

    // --- MLP chain, in place on the wave buffer ---
    mlp_layer<3, 8>(buf, wlds + T_WS0 * 512, bs0, lane);   // 96 ->128
    mlp_layer<4, 8>(buf, wlds + T_WS1 * 512, bs1, lane);   // 128->128
    mlp_layer<4, 8>(buf, wlds + T_WS2 * 512, bs2, lane);   // 128->128
    mlp_layer<6, 8>(buf, wlds + T_WT0 * 512, bt0, lane);   // 192->128 (h|tc)
    mlp_layer<4, 8>(buf, wlds + T_WT1 * 512, bt1, lane);   // 128->128
    head_layer(buf, wlds + T_HEAD * 512, bsh, bt2, maskp, out, npix, pix0, lane);
  }
}

// ---------------- host launcher ----------------
extern "C" void kernel_launch(void* const* d_in, const int* in_sizes, int n_in,
                              void* d_out, int out_size, void* d_ws, size_t ws_size,
                              hipStream_t stream) {
  const float* coords = (const float*)d_in[0];
  const float* shape_cond = (const float*)d_in[1];
  const float* tex_cond = (const float*)d_in[2];
  const float* ws0 = (const float*)d_in[3];
  const float* bs0 = (const float*)d_in[4];
  const float* ws1 = (const float*)d_in[5];
  const float* bs1 = (const float*)d_in[6];
  const float* ws2 = (const float*)d_in[7];
  const float* bs2 = (const float*)d_in[8];
  const float* wsh = (const float*)d_in[9];
  const float* bsh = (const float*)d_in[10];
  const float* wt0 = (const float*)d_in[11];
  const float* bt0 = (const float*)d_in[12];
  const float* wt1 = (const float*)d_in[13];
  const float* bt1 = (const float*)d_in[14];
  const float* wt2 = (const float*)d_in[15];
  const float* bt2 = (const float*)d_in[16];
  float* out = (float*)d_out;
  _Float16* packed = (_Float16*)d_ws;

  const int npix = in_sizes[0] / 2;      // 8*256*256 = 524288
  const int ntiles = npix / 16;          // 32768

  pack_weights_kernel<<<(PACKED_HALVES + 255) / 256, 256, 0, stream>>>(
      ws0, ws1, ws2, wsh, wt0, wt1, wt2, packed);

  (void)hipFuncSetAttribute((const void*)decoder_kernel,
                            hipFuncAttributeMaxDynamicSharedMemorySize,
                            SMEM_BYTES);

  decoder_kernel<<<512, 512, SMEM_BYTES, stream>>>(
      coords, shape_cond, tex_cond, packed,
      bs0, bs1, bs2, bsh, bt0, bt1, bt2,
      out, npix, ntiles);
}
